// CSMGC_13443247636664
// MI455X (gfx1250) — matrix operand order repaired and checked by reading, not compiled, not run on hardware
//
#include <hip/hip_runtime.h>

// ---------------------------------------------------------------------------
// Problem constants (match the reference)
// ---------------------------------------------------------------------------
#define BB    16
#define CC    128
#define NN    2000
#define C2    256
#define COUT  128
#define NT    125          // NN / 16
#define BN_EPS 1e-5f
#define IN_EPS 1e-3f

typedef __attribute__((ext_vector_type(16))) __bf16 v16bf;
typedef __attribute__((ext_vector_type(8)))  float  v8f;

// ---------------------------------------------------------------------------
// WMMA fragment helpers (CDNA5 wave32 layouts, CDNA5 ISA §7.12.2)
// A (16x32 bf16): lane L -> row = L&15; elems 0..7 -> K = hf*8 + i,
//                 elems 8..15 -> K = 16 + hf*8 + (i-8), hf = L>>4
// B (32x16 bf16): lane L -> col = L&15; same K order as A
// C/D (16x16 f32): lane L -> col N = L&15; vgpr j -> row M = hf*8 + j
// ---------------------------------------------------------------------------
__device__ __forceinline__ v16bf frag_a(const __bf16* p) { // p = row base + k0 + hf*8
  v16bf a;
#pragma unroll
  for (int i = 0; i < 8; ++i) { a[i] = p[i]; a[i + 8] = p[i + 16]; }
  return a;
}
__device__ __forceinline__ v16bf frag_b(const __bf16* p) { // p = col base + k0 + hf*8
  v16bf b;
#pragma unroll
  for (int i = 0; i < 8; ++i) { b[i] = p[i]; b[i + 8] = p[i + 16]; }
  return b;
}
__device__ __forceinline__ v8f wmma_bf16(v16bf a, v16bf b, v8f c) {
  return __builtin_amdgcn_wmma_f32_16x16x32_bf16(false, a, false, b, (short)0, c,
                                                 false, false);
}
__device__ __forceinline__ v8f vzero8() {
  v8f z = {0.f, 0.f, 0.f, 0.f, 0.f, 0.f, 0.f, 0.f};
  return z;
}

// ---------------------------------------------------------------------------
// Prep: transpose x [B,C,N] -> xT [row=(s*B+b)*N+n][C] bf16 ; row L2 norms
// ---------------------------------------------------------------------------
__global__ void k_transpose(const float* __restrict__ x, __bf16* __restrict__ xT) {
  int i = blockIdx.x * 256 + threadIdx.x;            // over B*C*N = 4.096M
  if (i >= BB * CC * NN) return;
  int b = i / (CC * NN);
  int r = i - b * CC * NN;
  int c = r / NN;
  int n = r - c * NN;
  xT[((size_t)(b * NN + n)) * CC + c] = (__bf16)x[i];
}

__global__ void k_norms(const __bf16* __restrict__ xT, float* __restrict__ xx) {
  int row = blockIdx.x * 256 + threadIdx.x;          // over 2*B*N = 64000
  if (row >= 2 * BB * NN) return;
  const __bf16* p = xT + (size_t)row * CC;
  float s = 0.f;
#pragma unroll 8
  for (int c = 0; c < CC; ++c) { float v = (float)p[c]; s += v * v; }
  xx[row] = s;
}

// ---------------------------------------------------------------------------
// Weight prep: fold W1 into Wcat[256][384]; W2 -> W2e[256][512]; W3 -> bf16
// (b1/b2 dropped: bias before training-mode BN cancels)
// ---------------------------------------------------------------------------
__global__ void k_wprep(const float* __restrict__ W1, const float* __restrict__ W2,
                        const float* __restrict__ W3, __bf16* __restrict__ Wcat,
                        __bf16* __restrict__ W2e, __bf16* __restrict__ W3b) {
  int i = blockIdx.x * 256 + threadIdx.x;
  if (i < 256 * 384) {
    int co = i / 384, k = i - co * 384;
    float v;
    if (k < 128) {
      v = W1[co * 512 + k * 2] + W1[co * 512 + k * 2 + 1] +
          W1[co * 512 + (k + 128) * 2] + W1[co * 512 + (k + 128) * 2 + 1];
    } else if (k < 256) {
      int c = k - 128; v = -W1[co * 512 + (c + 128) * 2];
    } else {
      int c = k - 256; v = -W1[co * 512 + (c + 128) * 2 + 1];
    }
    Wcat[i] = (__bf16)v;
  } else if (i < 256 * 384 + 256 * 512) {
    int j = i - 256 * 384;
    int co = j / 512, k = j - co * 512;
    int s = k >> 8, ci = k & 255;
    W2e[j] = (__bf16)W2[co * 512 + ci * 2 + s];
  } else if (i < 256 * 384 + 256 * 512 + COUT * 256) {
    int j = i - (256 * 384 + 256 * 512);
    W3b[j] = (__bf16)W3[j];
  }
}

// ---------------------------------------------------------------------------
// Fused KNN: WMMA Gram tiles + streaming argmax of (2*inner - ||x_j||^2), j!=i
// grid (125 row-tiles, B, 2 stages), block 256 (8 waves over column tiles)
// ---------------------------------------------------------------------------
__global__ void k_knn(const __bf16* __restrict__ xT, const float* __restrict__ xx,
                      int* __restrict__ nbr) {
  const int rt = blockIdx.x, b = blockIdx.y, s = blockIdx.z;
  const __bf16* X = xT + ((size_t)(s * BB + b)) * NN * CC;
  const float* XX = xx + (s * BB + b) * NN;
  const int lane = threadIdx.x & 31, wave = threadIdx.x >> 5;
  const int hf = lane >> 4, l15 = lane & 15;
  const int r0 = rt * 16;

  v16bf afr[4];
  const __bf16* arow = X + (size_t)(r0 + l15) * CC;
#pragma unroll
  for (int kk = 0; kk < 4; ++kk) afr[kk] = frag_a(arow + kk * 32 + hf * 8);

  float bestv[8]; int besti[8];
#pragma unroll
  for (int j = 0; j < 8; ++j) { bestv[j] = -3.0e38f; besti[j] = 0; }

  for (int ct = wave; ct < NT; ct += 8) {
    const int c0 = ct * 16;
    v8f acc = vzero8();
    const __bf16* bcol = X + (size_t)(c0 + l15) * CC;
#pragma unroll
    for (int kk = 0; kk < 4; ++kk)
      acc = wmma_bf16(afr[kk], frag_b(bcol + kk * 32 + hf * 8), acc);
    const int col = c0 + l15;
    const float xxc = XX[col];
#pragma unroll
    for (int j = 0; j < 8; ++j) {
      int row = r0 + hf * 8 + j;
      float cand = 2.f * acc[j] - xxc;
      if (col != row && cand > bestv[j]) { bestv[j] = cand; besti[j] = col; }
    }
  }
  // reduce across the 16 lanes holding the same row
#pragma unroll
  for (int m = 1; m < 16; m <<= 1) {
#pragma unroll
    for (int j = 0; j < 8; ++j) {
      float ov = __shfl_xor(bestv[j], m);
      int   oi = __shfl_xor(besti[j], m);
      if (ov > bestv[j]) { bestv[j] = ov; besti[j] = oi; }
    }
  }
  __shared__ float smv[8][16];
  __shared__ int   smi[8][16];
  if (l15 == 0) {
#pragma unroll
    for (int j = 0; j < 8; ++j) { smv[wave][hf * 8 + j] = bestv[j]; smi[wave][hf * 8 + j] = besti[j]; }
  }
  __syncthreads();
  if (threadIdx.x < 16) {
    float bv = smv[0][threadIdx.x]; int bi = smi[0][threadIdx.x];
#pragma unroll
    for (int w = 1; w < 8; ++w)
      if (smv[w][threadIdx.x] > bv) { bv = smv[w][threadIdx.x]; bi = smi[w][threadIdx.x]; }
    nbr[(s * BB + b) * NN + r0 + threadIdx.x] = bi;
  }
}

// ---------------------------------------------------------------------------
// Conv1 as WMMA GEMM: h1[b][n][s][256] = Wcat[256x384] @ [x_n; x_{i0=n}; x_{i1}]
// grid (125, B, 2), block 256; F-tile staged in LDS (gathered via nbr)
// ---------------------------------------------------------------------------
__global__ void k_conv1(const __bf16* __restrict__ xT, const int* __restrict__ nbr,
                        const __bf16* __restrict__ Wcat, float* __restrict__ h1) {
  __shared__ __bf16 Ft[16 * 392];                    // 384 + 8 pad per point
  const int pt = blockIdx.x, b = blockIdx.y, s = blockIdx.z;
  const int n0 = pt * 16;
  const __bf16* X = xT + ((size_t)(s * BB + b)) * NN * CC;
  const int* NB = nbr + (s * BB + b) * NN;

  for (int it = 0; it < 3; ++it) {                   // 768 16-byte chunks
    int ch = threadIdx.x + it * 256;
    int pr = ch / 48, cid = ch - pr * 48;
    int off = cid * 8;
    int seg = off >> 7, within = off & 127;
    int p = n0 + pr;
    int src = (seg == 2) ? NB[p] : p;
    const uint4* g = reinterpret_cast<const uint4*>(X + (size_t)src * CC + within);
    *reinterpret_cast<uint4*>(&Ft[pr * 392 + off]) = *g;
  }
  __syncthreads();

  const int lane = threadIdx.x & 31, wave = threadIdx.x >> 5;
  const int hf = lane >> 4, l15 = lane & 15;
  v8f acc0 = vzero8(), acc1 = vzero8();
  const __bf16* A0 = Wcat + (size_t)(wave * 16 + l15) * 384;
  const __bf16* A1 = Wcat + (size_t)((wave + 8) * 16 + l15) * 384;
  const __bf16* Bp = &Ft[l15 * 392];
#pragma unroll
  for (int kk = 0; kk < 12; ++kk) {
    int k0 = kk * 32;
    v16bf bf = frag_b(Bp + k0 + hf * 8);
    acc0 = wmma_bf16(frag_a(A0 + k0 + hf * 8), bf, acc0);
    acc1 = wmma_bf16(frag_a(A1 + k0 + hf * 8), bf, acc1);
  }
  float* out = h1 + ((size_t)((b * NN + n0 + l15) * 2 + s)) * 256;
#pragma unroll
  for (int j = 0; j < 8; ++j) out[wave * 16 + hf * 8 + j] = acc0[j];
#pragma unroll
  for (int j = 0; j < 8; ++j) out[(wave + 8) * 16 + hf * 8 + j] = acc1[j];
}

// ---------------------------------------------------------------------------
// Conv2: h2[b][n][256] = W2e[256x512] @ h1b[b][n][:]   (K = s*256+ci)
// ---------------------------------------------------------------------------
__global__ void k_conv2(const __bf16* __restrict__ h1b, const __bf16* __restrict__ W2e,
                        float* __restrict__ h2) {
  __shared__ __bf16 Ft[16 * 520];
  const int pt = blockIdx.x, b = blockIdx.y;
  const int n0 = pt * 16;
  for (int it = 0; it < 4; ++it) {                   // 1024 chunks
    int ch = threadIdx.x + it * 256;
    int pr = ch >> 6, off = (ch & 63) * 8;
    const uint4* g = reinterpret_cast<const uint4*>(
        h1b + ((size_t)(b * NN + n0 + pr)) * 512 + off);
    *reinterpret_cast<uint4*>(&Ft[pr * 520 + off]) = *g;
  }
  __syncthreads();
  const int lane = threadIdx.x & 31, wave = threadIdx.x >> 5;
  const int hf = lane >> 4, l15 = lane & 15;
  v8f acc0 = vzero8(), acc1 = vzero8();
  const __bf16* A0 = W2e + (size_t)(wave * 16 + l15) * 512;
  const __bf16* A1 = W2e + (size_t)((wave + 8) * 16 + l15) * 512;
  const __bf16* Bp = &Ft[l15 * 520];
#pragma unroll
  for (int kk = 0; kk < 16; ++kk) {
    int k0 = kk * 32;
    v16bf bf = frag_b(Bp + k0 + hf * 8);
    acc0 = wmma_bf16(frag_a(A0 + k0 + hf * 8), bf, acc0);
    acc1 = wmma_bf16(frag_a(A1 + k0 + hf * 8), bf, acc1);
  }
  float* out = h2 + ((size_t)(b * NN + n0 + l15)) * 256;
#pragma unroll
  for (int j = 0; j < 8; ++j) out[wave * 16 + hf * 8 + j] = acc0[j];
#pragma unroll
  for (int j = 0; j < 8; ++j) out[(wave + 8) * 16 + hf * 8 + j] = acc1[j];
}

// ---------------------------------------------------------------------------
// Conv3 + bias + residual: out[b][co][n] = W3b[128x256]@h3b + b3 + Stage_2
// ---------------------------------------------------------------------------
__global__ void k_conv3(const __bf16* __restrict__ h3b, const __bf16* __restrict__ W3b,
                        const float* __restrict__ b3, const float* __restrict__ s2,
                        float* __restrict__ out) {
  __shared__ __bf16 Ft[16 * 264];
  const int pt = blockIdx.x, b = blockIdx.y;
  const int n0 = pt * 16;
  for (int it = 0; it < 2; ++it) {                   // 512 chunks
    int ch = threadIdx.x + it * 256;
    int pr = ch >> 5, off = (ch & 31) * 8;
    const uint4* g = reinterpret_cast<const uint4*>(
        h3b + ((size_t)(b * NN + n0 + pr)) * 256 + off);
    *reinterpret_cast<uint4*>(&Ft[pr * 264 + off]) = *g;
  }
  __syncthreads();
  const int lane = threadIdx.x & 31, wave = threadIdx.x >> 5;
  const int hf = lane >> 4, l15 = lane & 15;
  v8f acc = vzero8();
  const __bf16* A0 = W3b + (size_t)(wave * 16 + l15) * 256;
  const __bf16* Bp = &Ft[l15 * 264];
#pragma unroll
  for (int kk = 0; kk < 8; ++kk) {
    int k0 = kk * 32;
    acc = wmma_bf16(frag_a(A0 + k0 + hf * 8), frag_b(Bp + k0 + hf * 8), acc);
  }
  const int p = n0 + l15;
#pragma unroll
  for (int j = 0; j < 8; ++j) {
    int co = wave * 16 + hf * 8 + j;
    size_t oi = ((size_t)(b * COUT + co)) * NN + p;
    out[oi] = acc[j] + b3[co] + s2[oi];
  }
}

// ---------------------------------------------------------------------------
// Normalization kernels
// ---------------------------------------------------------------------------
__global__ void k_zero(float* __restrict__ p, int n) {
  int i = blockIdx.x * 256 + threadIdx.x;
  if (i < n) p[i] = 0.f;
}

// per-channel sum / sumsq over rows of 256 channels (256 rows per block)
__global__ void k_stats(const float* __restrict__ x, float* __restrict__ S,
                        float* __restrict__ Q) {
  const int c = threadIdx.x;
  const float* base = x + ((size_t)blockIdx.x) * 256 * 256 + c;
  float s = 0.f, q = 0.f;
#pragma unroll 4
  for (int r = 0; r < 256; ++r) { float v = base[(size_t)r * 256]; s += v; q += v * v; }
  atomicAdd(&S[c], s);
  atomicAdd(&Q[c], q);
}

__global__ void k_fin(const float* __restrict__ S, const float* __restrict__ Q,
                      const float* __restrict__ g, float* __restrict__ m,
                      float* __restrict__ rg, float inv_cnt) {
  int t = threadIdx.x;
  float mean = S[t] * inv_cnt;
  float var = Q[t] * inv_cnt - mean * mean;
  m[t] = mean;
  rg[t] = rsqrtf(var + BN_EPS) * g[t];
}

__global__ void k_apply_bf(const float* __restrict__ x, const float* __restrict__ m,
                           const float* __restrict__ rg, const float* __restrict__ be,
                           __bf16* __restrict__ y, int total) {
  int i = blockIdx.x * 256 + threadIdx.x;
  if (i >= total) return;
  int c = i & 255;
  float v = (x[i] - m[c]) * rg[c] + be[c];
  y[i] = (__bf16)(v > 0.f ? v : 0.f);
}

__global__ void k_apply_f(const float* __restrict__ x, const float* __restrict__ m,
                          const float* __restrict__ rg, const float* __restrict__ be,
                          float* __restrict__ y, int total) {
  int i = blockIdx.x * 256 + threadIdx.x;
  if (i >= total) return;
  int c = i & 255;
  float v = (x[i] - m[c]) * rg[c] + be[c];
  y[i] = v > 0.f ? v : 0.f;
}

// instance-norm stats: grid (B, 8), 250 rows per block
__global__ void k_instats(const float* __restrict__ h2, float* __restrict__ Sin,
                          float* __restrict__ Qin) {
  const int b = blockIdx.x, ch = blockIdx.y, c = threadIdx.x;
  const float* base = h2 + (((size_t)b * NN) + ch * 250) * 256 + c;
  float s = 0.f, q = 0.f;
  for (int r = 0; r < 250; ++r) { float v = base[(size_t)r * 256]; s += v; q += v * v; }
  atomicAdd(&Sin[b * 256 + c], s);
  atomicAdd(&Qin[b * 256 + c], q);
}

// finalize IN stats; accumulate BN3 variance = mean_b[v/(v+IN_EPS)] (mean is 0)
__global__ void k_finin(const float* __restrict__ Sin, const float* __restrict__ Qin,
                        float* __restrict__ mIn, float* __restrict__ rIn,
                        float* __restrict__ Vc) {
  const int b = blockIdx.x, t = threadIdx.x, i = b * 256 + t;
  float mean = Sin[i] * (1.f / NN);
  float var = Qin[i] * (1.f / NN) - mean * mean;
  mIn[i] = mean;
  rIn[i] = rsqrtf(var + IN_EPS);
  atomicAdd(&Vc[t], var / (var + IN_EPS));
}

__global__ void k_fin3(const float* __restrict__ Vc, const float* __restrict__ g3,
                       float* __restrict__ sc3) {
  int t = threadIdx.x;
  sc3[t] = rsqrtf(Vc[t] * (1.f / BB) + BN_EPS) * g3[t];
}

// fused IN + BN3 + ReLU -> bf16
__global__ void k_apply3(const float* __restrict__ h2, const float* __restrict__ mIn,
                         const float* __restrict__ rIn, const float* __restrict__ sc3,
                         const float* __restrict__ be3, __bf16* __restrict__ h3b,
                         int total) {
  int i = blockIdx.x * 256 + threadIdx.x;
  if (i >= total) return;
  int c = i & 255;
  int b = i / (NN * 256);
  float v = (h2[i] - mIn[b * 256 + c]) * rIn[b * 256 + c] * sc3[c] + be3[c];
  h3b[i] = (__bf16)(v > 0.f ? v : 0.f);
}

// ---------------------------------------------------------------------------
// Host launcher
// ---------------------------------------------------------------------------
extern "C" void kernel_launch(void* const* d_in, const int* in_sizes, int n_in,
                              void* d_out, int out_size, void* d_ws, size_t ws_size,
                              hipStream_t stream) {
  const float* Stage1 = (const float*)d_in[0];
  const float* Stage2 = (const float*)d_in[1];
  const float* W1 = (const float*)d_in[2];
  const float* g1 = (const float*)d_in[4];
  const float* be1 = (const float*)d_in[5];
  const float* W2 = (const float*)d_in[6];
  const float* g2 = (const float*)d_in[8];
  const float* be2 = (const float*)d_in[9];
  const float* g3 = (const float*)d_in[10];
  const float* be3 = (const float*)d_in[11];
  const float* W3 = (const float*)d_in[12];
  const float* b3 = (const float*)d_in[13];
  float* out = (float*)d_out;

  // ---- workspace layout (~149 MB), 256-byte aligned ----
  char* ws = (char*)d_ws;
  size_t o = 0;
  auto alloc = [&](size_t bytes) {
    size_t r = o; o = (o + bytes + 255) & ~(size_t)255; return r;
  };
  __bf16* xT   = (__bf16*)(ws + alloc((size_t)2 * BB * NN * CC * 2)); // 16.4 MB
  float*  xx   = (float*)(ws + alloc((size_t)2 * BB * NN * 4));
  int*    nbr  = (int*)(ws + alloc((size_t)2 * BB * NN * 4));
  __bf16* Wcat = (__bf16*)(ws + alloc((size_t)256 * 384 * 2));
  __bf16* W2e  = (__bf16*)(ws + alloc((size_t)256 * 512 * 2));
  __bf16* W3b  = (__bf16*)(ws + alloc((size_t)COUT * 256 * 2));
  float*  h1   = (float*)(ws + alloc((size_t)BB * NN * 2 * 256 * 4));  // 65.5 MB
  __bf16* h1b  = (__bf16*)(ws + alloc((size_t)BB * NN * 2 * 256 * 2)); // 32.8 MB
  float*  h2   = (float*)(ws + alloc((size_t)BB * NN * 256 * 4));      // 32.8 MB
  float*  S1   = (float*)(ws + alloc(256 * 4));
  float*  Q1   = (float*)(ws + alloc(256 * 4));
  float*  m1   = (float*)(ws + alloc(256 * 4));
  float*  r1   = (float*)(ws + alloc(256 * 4));
  float*  S2   = (float*)(ws + alloc(256 * 4));
  float*  Q2   = (float*)(ws + alloc(256 * 4));
  float*  m2   = (float*)(ws + alloc(256 * 4));
  float*  r2   = (float*)(ws + alloc(256 * 4));
  float*  Sin  = (float*)(ws + alloc((size_t)BB * 256 * 4));
  float*  Qin  = (float*)(ws + alloc((size_t)BB * 256 * 4));
  float*  mIn  = (float*)(ws + alloc((size_t)BB * 256 * 4));
  float*  rIn  = (float*)(ws + alloc((size_t)BB * 256 * 4));
  float*  Vc   = (float*)(ws + alloc(256 * 4));
  float*  sc3  = (float*)(ws + alloc(256 * 4));
  __bf16* h3b  = (__bf16*)h1;   // reuse h1 region (free after apply1)
  (void)in_sizes; (void)n_in; (void)out_size; (void)ws_size;

  // zero the atomic accumulators (S1..Vc region is contiguous)
  {
    int zn = (int)((Vc + 256) - S1);
    k_zero<<<(zn + 255) / 256, 256, 0, stream>>>(S1, zn);
  }

  // 1) transpose + norms + weight prep
  const int PERSTAGE = BB * CC * NN;
  k_transpose<<<(PERSTAGE + 255) / 256, 256, 0, stream>>>(Stage1, xT);
  k_transpose<<<(PERSTAGE + 255) / 256, 256, 0, stream>>>(Stage2,
                                                          xT + (size_t)BB * NN * CC);
  k_norms<<<(2 * BB * NN + 255) / 256, 256, 0, stream>>>(xT, xx);
  k_wprep<<<(256 * 384 + 256 * 512 + COUT * 256 + 255) / 256, 256, 0, stream>>>(
      W1, W2, W3, Wcat, W2e, W3b);

  // 2) fused KNN (WMMA Gram + argmax)
  k_knn<<<dim3(NT, BB, 2), 256, 0, stream>>>(xT, xx, nbr);

  // 3) conv1 -> BN1 -> ReLU
  k_conv1<<<dim3(NT, BB, 2), 256, 0, stream>>>(xT, nbr, Wcat, h1);
  k_stats<<<250, 256, 0, stream>>>(h1, S1, Q1);                 // 64000 rows
  k_fin<<<1, 256, 0, stream>>>(S1, Q1, g1, m1, r1, 1.f / (BB * NN * 2));
  k_apply_bf<<<(BB * NN * 2 * 256 + 255) / 256, 256, 0, stream>>>(
      h1, m1, r1, be1, h1b, BB * NN * 2 * 256);

  // 4) conv2 -> BN2 -> ReLU (in place)
  k_conv2<<<dim3(NT, BB), 256, 0, stream>>>(h1b, W2e, h2);
  k_stats<<<125, 256, 0, stream>>>(h2, S2, Q2);                 // 32000 rows
  k_fin<<<1, 256, 0, stream>>>(S2, Q2, g2, m2, r2, 1.f / (BB * NN));
  k_apply_f<<<(BB * NN * 256 + 255) / 256, 256, 0, stream>>>(
      h2, m2, r2, be2, h2, BB * NN * 256);

  // 5) InstanceNorm + BN3 (analytic) + ReLU -> bf16
  k_instats<<<dim3(BB, 8), 256, 0, stream>>>(h2, Sin, Qin);
  k_finin<<<BB, 256, 0, stream>>>(Sin, Qin, mIn, rIn, Vc);
  k_fin3<<<1, 256, 0, stream>>>(Vc, g3, sc3);
  k_apply3<<<(BB * NN * 256 + 255) / 256, 256, 0, stream>>>(
      h2, mIn, rIn, sc3, be3, h3b, BB * NN * 256);

  // 6) conv3 + b3 + residual
  k_conv3<<<dim3(NT, BB), 256, 0, stream>>>(h3b, W3b, b3, Stage2, out);
}
